// EncoderSelfMSA_41180146434240
// MI455X (gfx1250) — matrix operand order repaired and checked by reading, not compiled, not run on hardware
//
#include <hip/hip_runtime.h>
#include <math.h>

// ---------------------------------------------------------------------------
// Swin-style shifted-window MSA for MI455X (gfx1250), bf16 WMMA everywhere.
// E=256, NH=8, HD=32, WS=8, WW=64, HG=128, N=16384, B=4, NWIN=16.
// ---------------------------------------------------------------------------

typedef __attribute__((ext_vector_type(16))) __bf16 v16bf;
typedef __attribute__((ext_vector_type(8)))  float  v8f;

union AF { v16bf v; unsigned int u[8]; };
union CF { v8f  v; float f[8]; };

__device__ inline unsigned short f2bf(float f) {
  unsigned int x = __float_as_uint(f);
  // round-to-nearest-even truncation to bf16
  return (unsigned short)((x + 0x7FFFu + ((x >> 16) & 1u)) >> 16);
}

// A-matrix fragment (16x32 bf16), ISA layout:
// lane = m (&15); half=lane>>4; VGPR i<4: k=2i+8*half ; i>=4: k=2i+8+8*half
__device__ inline void load_afrag(const unsigned short* lds, int ld,
                                  int mBase, int kBase, int lane, AF& f) {
  const int m = mBase + (lane & 15);
  const int h8 = (lane >> 4) << 3;
  const unsigned short* row = lds + m * ld + kBase;
#pragma unroll
  for (int i = 0; i < 8; ++i) {
    const int k = (i < 4) ? (2 * i + h8) : (2 * i + 8 + h8);
    f.u[i] = *(const unsigned int*)(row + k);
  }
}

// B-matrix fragment (32x16 bf16) read from LDS stored as [n][k]:
// lane = n (&15); half=lane>>4; VGPR i<4: k=2i+8*half ; i>=4: k=2i+8+8*half
__device__ inline void load_bfrag(const unsigned short* ldsNK, int ld,
                                  int nBase, int kBase, int lane, AF& f) {
  const int n = nBase + (lane & 15);
  const unsigned short* row = ldsNK + n * ld + kBase + ((lane >> 4) << 3);
#pragma unroll
  for (int i = 0; i < 8; ++i)
    f.u[i] = *(const unsigned int*)(row + ((i < 4) ? (2 * i) : (2 * i + 8)));
}

__device__ inline v8f wmma_bf(const AF& a, const AF& b, v8f c) {
  return __builtin_amdgcn_wmma_f32_16x16x32_bf16(
      false, a.v, false, b.v, (short)0, c, false, false);
}

// ---------------------------------------------------------------------------
// Generic tiled GEMM: C[M,Nw] = A[M,K] @ W[K,Nw] + bias.
// AMODE: 0 = f32 A (convert to bf16 while staging), 1 = bf16 A.
// OMODE: 0 = bf16 out, 1 = f32 out.
// PERM : 0 = identity columns; 1 = QKV store (c=3e+g -> [g][h][hd]);
//        2 = sc/sh store (e -> [h][hd]), where e = hd*8 + h (reference split).
// Tiles: BM=128, BN=128, BK=32; 256 threads = 8 waves (4 m-waves x 2 n-waves),
// each wave owns a 32x64 tile = 2x4 WMMA accumulators.
// ---------------------------------------------------------------------------
template <int AMODE, int OMODE, int PERM>
__global__ __launch_bounds__(256) void wmma_gemm(
    const void* __restrict__ Aptr, const float* __restrict__ W,
    const float* __restrict__ bias, void* __restrict__ Cptr,
    int M, int K, int Nw) {
  __shared__ unsigned short As[128 * 34];   // [m][k]
  __shared__ unsigned short Bs[128 * 34];   // [n][k]

  const int tid  = threadIdx.x;
  const int lane = tid & 31;
  const int wave = tid >> 5;
  const int wm = wave >> 1;            // 0..3
  const int wn = wave & 1;             // 0..1
  const int m0 = blockIdx.x * 128;
  const int n0 = blockIdx.y * 128;

  CF acc[2][4];
#pragma unroll
  for (int i = 0; i < 2; ++i)
#pragma unroll
    for (int j = 0; j < 4; ++j)
#pragma unroll
      for (int r = 0; r < 8; ++r) acc[i][j].f[r] = 0.0f;

  for (int k0 = 0; k0 < K; k0 += 32) {
    __syncthreads();
    // ---- stage A tile (128x32) ----
    {
      const int rbase = tid >> 3;            // 0..31
      const int c = (tid & 7) * 4;           // 0..28
#pragma unroll
      for (int rr = 0; rr < 4; ++rr) {
        const int row = rbase + rr * 32;
        unsigned int* dst = (unsigned int*)&As[row * 34 + c];
        if (AMODE == 0) {
          const float* A = (const float*)Aptr;
          float4 fv = *(const float4*)(A + (size_t)(m0 + row) * K + k0 + c);
          dst[0] = (unsigned int)f2bf(fv.x) | ((unsigned int)f2bf(fv.y) << 16);
          dst[1] = (unsigned int)f2bf(fv.z) | ((unsigned int)f2bf(fv.w) << 16);
        } else {
          const unsigned short* A = (const unsigned short*)Aptr;
          uint2 pv = *(const uint2*)(A + (size_t)(m0 + row) * K + k0 + c);
          dst[0] = pv.x; dst[1] = pv.y;
        }
      }
    }
    // ---- stage W tile (32x128) transposed into Bs[n][k] ----
    {
      const int kb = tid >> 5;               // 0..7
      const int c = lane * 4;                // 0..124
#pragma unroll
      for (int rr = 0; rr < 4; ++rr) {
        const int kk = kb + rr * 8;
        float4 fv = *(const float4*)(W + (size_t)(k0 + kk) * Nw + n0 + c);
        Bs[(c + 0) * 34 + kk] = f2bf(fv.x);
        Bs[(c + 1) * 34 + kk] = f2bf(fv.y);
        Bs[(c + 2) * 34 + kk] = f2bf(fv.z);
        Bs[(c + 3) * 34 + kk] = f2bf(fv.w);
      }
    }
    __syncthreads();

    AF a[2], b[4];
#pragma unroll
    for (int i = 0; i < 2; ++i) load_afrag(As, 34, wm * 32 + i * 16, 0, lane, a[i]);
#pragma unroll
    for (int j = 0; j < 4; ++j) load_bfrag(Bs, 34, wn * 64 + j * 16, 0, lane, b[j]);
#pragma unroll
    for (int i = 0; i < 2; ++i)
#pragma unroll
      for (int j = 0; j < 4; ++j)
        acc[i][j].v = wmma_bf(a[i], b[j], acc[i][j].v);
  }

  // ---- epilogue ----
  const int half = lane >> 4;
  const int nl = lane & 15;
#pragma unroll
  for (int i = 0; i < 2; ++i)
#pragma unroll
    for (int j = 0; j < 4; ++j) {
      const int ng = n0 + wn * 64 + j * 16 + nl;
      const float bv = bias[ng];
      int col = ng;
      if (PERM == 1) { const int g = ng % 3, e = ng / 3; col = g * 256 + (e & 7) * 32 + (e >> 3); }
      else if (PERM == 2) { col = (ng & 7) * 32 + (ng >> 3); }
#pragma unroll
      for (int r = 0; r < 8; ++r) {
        const int mg = m0 + wm * 32 + i * 16 + r + 8 * half;
        const float val = acc[i][j].f[r] + bv;
        if (OMODE == 0)
          ((unsigned short*)Cptr)[(size_t)mg * Nw + col] = f2bf(val);
        else
          ((float*)Cptr)[(size_t)mg * Nw + col] = val;
      }
    }
}

// ---------------------------------------------------------------------------
// Windowed attention: one 128-thread block (4 waves) per (b, h, u, v) window.
// Wave w owns the 16-query strip [16w, 16w+16). HD=32 => QK^T is ONE wmma per
// 16x16 tile; w@V is 2 k-steps. Softmax in f32 with shfl_xor row reductions.
// ---------------------------------------------------------------------------
__global__ __launch_bounds__(128) void attn_kernel(
    const unsigned short* __restrict__ qkv,  // [B*N][3][256] bf16, [g][h][hd]
    const unsigned short* __restrict__ scb,  // [B*N][256] bf16, [h][hd]
    const unsigned short* __restrict__ shb,  // [B*N][256] bf16, [h][hd]
    const float* __restrict__ pos_emb,       // 15x15 f32
    unsigned short* __restrict__ o0,         // merged outputs, bf16 [B*N][256]
    unsigned short* __restrict__ o1,
    unsigned short* __restrict__ o2) {
  __shared__ unsigned short Qs[64 * 34];     // [p][hd]
  __shared__ unsigned short Ks[64 * 34];     // [q][hd]   (== B[n][k] for QK^T)
  __shared__ unsigned short Vt[3][32 * 66];  // [hd][q]   (== B[n][k] for w@V)
  __shared__ unsigned short Ps[64 * 66];     // probs [p][q], bf16
  __shared__ float pe[226];

  const int wid = blockIdx.x;
  const int v = wid & 15, u = (wid >> 4) & 15, h = (wid >> 8) & 7, b = wid >> 11;
  const int tid = threadIdx.x, lane = tid & 31, wave = tid >> 5;

  for (int i = tid; i < 225; i += 128) pe[i] = pos_emb[i];

  // ---- gather rolled window rows into LDS (64B contiguous per row) ----
  const int rowIdx = tid >> 4;   // 0..7
  const int cw = tid & 15;       // dword column: hd pair {2cw, 2cw+1}
#pragma unroll
  for (int p0 = 0; p0 < 64; p0 += 8) {
    const int p = p0 + rowIdx;
    const int ir = u * 8 + (p >> 3), jr = v * 8 + (p & 7);
    const int io = (ir + 4) & 127, jo = (jr + 4) & 127;   // un-roll(-4,-4)
    const size_t pos = (size_t)b * 16384 + (size_t)io * 128 + jo;
    const unsigned int* qr = (const unsigned int*)(qkv + (pos * 3 + 0) * 256 + h * 32);
    const unsigned int* kr = (const unsigned int*)(qkv + (pos * 3 + 1) * 256 + h * 32);
    const unsigned int* vr = (const unsigned int*)(qkv + (pos * 3 + 2) * 256 + h * 32);
    const unsigned int* sr = (const unsigned int*)(scb + pos * 256 + h * 32);
    const unsigned int* hr = (const unsigned int*)(shb + pos * 256 + h * 32);
    *(unsigned int*)&Qs[p * 34 + cw * 2] = qr[cw];
    *(unsigned int*)&Ks[p * 34 + cw * 2] = kr[cw];
    const unsigned int vv = vr[cw], sv = sr[cw], hv = hr[cw];
    const int hd = cw * 2;
    Vt[0][hd * 66 + p] = (unsigned short)vv; Vt[0][(hd + 1) * 66 + p] = (unsigned short)(vv >> 16);
    Vt[1][hd * 66 + p] = (unsigned short)sv; Vt[1][(hd + 1) * 66 + p] = (unsigned short)(sv >> 16);
    Vt[2][hd * 66 + p] = (unsigned short)hv; Vt[2][(hd + 1) * 66 + p] = (unsigned short)(hv >> 16);
  }
  __syncthreads();

  // ---- QK^T: 16x64 strip, 4 wmma ----
  const int pBase = wave * 16;
  AF aq;
  load_afrag(Qs, 34, pBase, 0, lane, aq);
  CF wacc[4];
#pragma unroll
  for (int nt = 0; nt < 4; ++nt) {
    AF bk;
    load_bfrag(Ks, 34, nt * 16, 0, lane, bk);
    v8f z = {};
    wacc[nt].v = wmma_bf(aq, bk, z);
  }

  // ---- bias + shifted-window masks + softmax (f32), write probs bf16 ----
  const float qscale = 0.17677669529663687f;  // 1/sqrt(HD)
  const int half = lane >> 4, nl = lane & 15;
#pragma unroll
  for (int r = 0; r < 8; ++r) {
    const int m = pBase + r + 8 * half;
    const int pa = m >> 3, pb = m & 7;
    float wv[4];
    float mx = -3.0e38f;
#pragma unroll
    for (int nt = 0; nt < 4; ++nt) {
      const int q = nt * 16 + nl;
      const int qa = q >> 3, qb = q & 7;
      float x = wacc[nt].f[r] * qscale + pe[(qa - pa + 7) * 15 + (qb - pb + 7)];
      if (u == 15 && ((pa >= 4) != (qa >= 4))) x = -1.0e30f;
      if (v == 15 && ((pb >= 4) != (qb >= 4))) x = -1.0e30f;
      wv[nt] = x;
      mx = fmaxf(mx, x);
    }
#pragma unroll
    for (int off = 1; off < 16; off <<= 1) mx = fmaxf(mx, __shfl_xor(mx, off, 32));
    float s = 0.0f;
#pragma unroll
    for (int nt = 0; nt < 4; ++nt) { wv[nt] = __expf(wv[nt] - mx); s += wv[nt]; }
#pragma unroll
    for (int off = 1; off < 16; off <<= 1) s += __shfl_xor(s, off, 32);
    const float inv = 1.0f / s;
#pragma unroll
    for (int nt = 0; nt < 4; ++nt)
      Ps[m * 66 + nt * 16 + nl] = f2bf(wv[nt] * inv);
  }
  __syncthreads();

  // ---- w @ {V, Vs, Vh}: 16x32 each, K=64 in 2 steps ----
  AF ap[2];
  load_afrag(Ps, 66, pBase, 0, lane, ap[0]);
  load_afrag(Ps, 66, pBase, 32, lane, ap[1]);
  CF oacc[3][2];
#pragma unroll
  for (int vm = 0; vm < 3; ++vm)
#pragma unroll
    for (int nt = 0; nt < 2; ++nt)
#pragma unroll
      for (int r = 0; r < 8; ++r) oacc[vm][nt].f[r] = 0.0f;

#pragma unroll
  for (int ks = 0; ks < 2; ++ks)
#pragma unroll
    for (int vm = 0; vm < 3; ++vm)
#pragma unroll
      for (int nt = 0; nt < 2; ++nt) {
        AF bv;
        load_bfrag(Vt[vm], 66, nt * 16, ks * 32, lane, bv);
        oacc[vm][nt].v = wmma_bf(ap[ks], bv, oacc[vm][nt].v);
      }

  // ---- merged store: n = (u*8+pi)*128 + v*8+pj, channel = h*32 + hd ----
  unsigned short* outs[3] = {o0, o1, o2};
#pragma unroll
  for (int vm = 0; vm < 3; ++vm)
#pragma unroll
    for (int nt = 0; nt < 2; ++nt)
#pragma unroll
      for (int r = 0; r < 8; ++r) {
        const int m = pBase + r + 8 * half;
        const int hd = nt * 16 + nl;
        const int ir = u * 8 + (m >> 3), jr = v * 8 + (m & 7);
        const size_t nidx = (size_t)b * 16384 + (size_t)ir * 128 + jr;
        outs[vm][nidx * 256 + h * 32 + hd] = f2bf(oacc[vm][nt].f[r]);
      }
}

// ---------------------------------------------------------------------------
extern "C" void kernel_launch(void* const* d_in, const int* in_sizes, int n_in,
                              void* d_out, int out_size, void* d_ws, size_t ws_size,
                              hipStream_t stream) {
  const float* x     = (const float*)d_in[0];
  const float* scale = (const float*)d_in[1];
  const float* shift = (const float*)d_in[2];
  const float* W1    = (const float*)d_in[3];
  const float* b1    = (const float*)d_in[4];
  const float* W1s   = (const float*)d_in[5];
  const float* b1s   = (const float*)d_in[6];
  const float* W1h   = (const float*)d_in[7];
  const float* b1h   = (const float*)d_in[8];
  const float* W2    = (const float*)d_in[9];
  const float* b2    = (const float*)d_in[10];
  const float* W2s   = (const float*)d_in[11];
  const float* b2s   = (const float*)d_in[12];
  const float* W2h   = (const float*)d_in[13];
  const float* b2h   = (const float*)d_in[14];
  const float* pemb  = (const float*)d_in[15];

  const size_t BN = 65536;  // B * N
  unsigned short* qkvb = (unsigned short*)d_ws;    // BN*768 bf16
  unsigned short* scbw = qkvb + BN * 768;          // BN*256
  unsigned short* shbw = scbw + BN * 256;          // BN*256
  unsigned short* at0  = shbw + BN * 256;          // BN*256 x3 merged attn
  unsigned short* at1  = at0 + BN * 256;
  unsigned short* at2  = at1 + BN * 256;           // total: 256 MB

  dim3 blk(256);
  // Stage 1: projections (f32 in -> bf16 staged, permuted for stage 2)
  wmma_gemm<0, 0, 1><<<dim3(512, 6), blk, 0, stream>>>(x, W1, b1, qkvb, 65536, 256, 768);
  wmma_gemm<0, 0, 2><<<dim3(512, 2), blk, 0, stream>>>(scale, W1s, b1s, scbw, 65536, 256, 256);
  wmma_gemm<0, 0, 2><<<dim3(512, 2), blk, 0, stream>>>(shift, W1h, b1h, shbw, 65536, 256, 256);

  // Stage 2: 8192 shifted windows, 4 waves each
  attn_kernel<<<8192, 128, 0, stream>>>(qkvb, scbw, shbw, pemb, at0, at1, at2);

  // Stage 3: output projections (bf16 in -> f32 out)
  float* out = (float*)d_out;
  const size_t BNE = BN * 256;
  wmma_gemm<1, 1, 0><<<dim3(512, 2), blk, 0, stream>>>(at0, W2,  b2,  out,           65536, 256, 256);
  wmma_gemm<1, 1, 0><<<dim3(512, 2), blk, 0, stream>>>(at1, W2s, b2s, out + BNE,     65536, 256, 256);
  wmma_gemm<1, 1, 0><<<dim3(512, 2), blk, 0, stream>>>(at2, W2h, b2h, out + 2 * BNE, 65536, 256, 256);
}
